// LSTMClassifier_29626684408121
// MI455X (gfx1250) — compile-verified, matrix operand order checked by reading, not run on hardware
//
#include <hip/hip_runtime.h>

// ---------------------------------------------------------------------------
// CDNA5 (gfx1250) LSTM classifier, bf16 WMMA pipeline.
//   B=64, T=3072, F_IN=128, CONV_OUT=64, K=3, STRIDE=3 -> T'=1024, HID=256, OUT=10
// ---------------------------------------------------------------------------

typedef __attribute__((ext_vector_type(16))) __bf16 v16bf;
typedef __attribute__((ext_vector_type(8)))  float  v8f;

union Frag {
    uint4   q[2];
    v16bf   v;
    __bf16  h[16];
};

#define LOG2E 1.4426950408889634f

// Short transcendental forms: v_exp_f32 / v_rcp_f32 are TRANS ops that
// co-execute with the XDL WMMA pipe.
__device__ __forceinline__ float fast_sigmoid(float x) {
    return __builtin_amdgcn_rcpf(1.f + __builtin_amdgcn_exp2f(-LOG2E * x));
}
__device__ __forceinline__ float fast_tanh(float x) {
    float e = __builtin_amdgcn_exp2f(-2.f * LOG2E * x);
    return (1.f - e) * __builtin_amdgcn_rcpf(1.f + e);
}

// ISA 16-bit A-fragment K index for element e (0..15) of a lane:
//   vgpr = e/2, half = e%2; K = (e>=8?16:0) + e%8, +8 for lanes 16..31.
__device__ __forceinline__ int kmap(int e, int laneHi) {
    return (e >= 8 ? 16 : 0) + (e & 7) + (laneHi ? 8 : 0);
}

// ---------------------------------------------------------------------------
// Kernel 1: inv L2 norm over time per (b, f).  grid=64, block=(128,8)
// ---------------------------------------------------------------------------
__global__ void k_norm(const float* __restrict__ in, float* __restrict__ invn) {
    __shared__ float part[8][128];
    int b = blockIdx.x, f = threadIdx.x, ty = threadIdx.y;
    const float* p = in + (size_t)b * 393216 + f;     // 3072*128
    float s = 0.f;
    for (int t = ty; t < 3072; t += 8) { float v = p[t * 128]; s += v * v; }
    part[ty][f] = s;
    __syncthreads();
    if (ty == 0) {
        float tot = 0.f;
        #pragma unroll
        for (int i = 0; i < 8; i++) tot += part[i][f];
        invn[b * 128 + f] = 1.f / fmaxf(sqrtf(tot), 1e-12f);
    }
}

// ---------------------------------------------------------------------------
// Kernel 2a: repack conv weights into WMMA B-fragment layout (bf16).
//   convpack[((ks*4+nt)*32+lane)*16+e],  ks<12, nt<4
//   n = nt*16+lane%16 (=o), K = ks*32+kmap(e,lane>=16), f=K%128, k=K/128
// ---------------------------------------------------------------------------
__global__ void k_pack_conv(const float* __restrict__ cw, __bf16* __restrict__ cp) {
    int idx = blockIdx.x * 256 + threadIdx.x;
    if (idx >= 12 * 4 * 32 * 16) return;
    int e = idx & 15, lane = (idx >> 4) & 31, nt = (idx >> 9) & 3, ks = idx >> 11;
    int o = nt * 16 + (lane & 15);
    int K = ks * 32 + kmap(e, lane >= 16);
    int f = K & 127, k = K >> 7;
    cp[idx] = (__bf16)cw[o * 384 + f * 3 + k];        // conv_w[o][f][k]
}

// ---------------------------------------------------------------------------
// Kernel 2b: repack [w_ih ; w_hh] (K=320, N=1024) into B-fragment layout.
//   wpack[((nt*10+ks)*32+lane)*16+e], nt<64, ks<10
// ---------------------------------------------------------------------------
__global__ void k_pack_w(const float* __restrict__ wih, const float* __restrict__ whh,
                         __bf16* __restrict__ wp) {
    int idx = blockIdx.x * 256 + threadIdx.x;
    if (idx >= 64 * 10 * 32 * 16) return;
    int e = idx & 15, lane = (idx >> 4) & 31;
    int ks = (idx / 512) % 10, nt = idx / 5120;
    int n = nt * 16 + (lane & 15);
    int K = ks * 32 + kmap(e, lane >= 16);
    wp[idx] = (__bf16)(K < 64 ? wih[n * 64 + K] : whh[n * 256 + (K - 64)]);
}

// ---------------------------------------------------------------------------
// Kernel 3: normalize + conv-as-GEMM + ReLU, output y in LSTM A-frag layout.
//   Per block: one batch b, 64 t' rows. 4 waves, wave = one 16-row M tile.
//   Conv bias enters through the C operand of the first WMMA.
//   yfrag[(((t'*2+ks)*4 + b/16)*32 + lane)*16 + e]  (K index = o, 0..63)
// ---------------------------------------------------------------------------
__global__ void __launch_bounds__(128) k_conv(
    const float* __restrict__ in, const float* __restrict__ invn,
    const float* __restrict__ convb, const __bf16* __restrict__ cp,
    __bf16* __restrict__ yfrag) {
    __shared__ float sinv[128];
    int b = blockIdx.y, tile = blockIdx.x, tid = threadIdx.x;
    int w = tid >> 5, L = tid & 31, laneHi = (L >= 16);
    if (tid < 128) sinv[tid] = invn[b * 128 + tid];
    __syncthreads();

    int t0 = tile * 64 + w * 16;
    v8f acc[4];
    const float* arow = in + (size_t)b * 393216;
    int r = t0 + (L & 15);

    #pragma unroll
    for (int ks = 0; ks < 12; ks++) {
        Frag a;
        #pragma unroll
        for (int e = 0; e < 16; e++) {
            int K = ks * 32 + kmap(e, laneHi);
            a.h[e] = (__bf16)(arow[r * 384 + K] * sinv[K & 127]);
        }
        #pragma unroll
        for (int nt = 0; nt < 4; nt++) {
            Frag bf;
            const uint4* src = (const uint4*)(cp + ((ks * 4 + nt) * 32 + L) * 16);
            bf.q[0] = src[0]; bf.q[1] = src[1];
            if (ks == 0) {
                v8f cb;
                float bv = convb[nt * 16 + (L & 15)];
                #pragma unroll
                for (int i = 0; i < 8; i++) cb[i] = bv;
                acc[nt] = __builtin_amdgcn_wmma_f32_16x16x32_bf16(
                    false, a.v, false, bf.v, (short)0, cb, false, false);
            } else {
                acc[nt] = __builtin_amdgcn_wmma_f32_16x16x32_bf16(
                    false, a.v, false, bf.v, (short)0, acc[nt], false, false);
            }
        }
    }
    // ReLU + scatter into LSTM A-fragment layout (rows become batch dim).
    #pragma unroll
    for (int nt = 0; nt < 4; nt++) {
        #pragma unroll
        for (int rg = 0; rg < 8; rg++) {
            float v  = fmaxf(acc[nt][rg], 0.f);
            int tp   = t0 + rg + (laneHi ? 8 : 0);
            int o    = nt * 16 + (L & 15);
            int ks2  = o >> 5;
            int hiN  = ((o & 15) >= 8);
            int Kl   = (o & 31) - (hiN ? 8 : 0);
            int e    = (Kl >= 16) ? (8 + Kl - 16) : Kl;
            int lane2 = hiN * 16 + (b & 15);
            size_t idx = ((((size_t)tp * 2 + ks2) * 4 + (b >> 4)) * 32 + lane2) * 16 + e;
            yfrag[idx] = (__bf16)v;
        }
    }
}

// ---------------------------------------------------------------------------
// Kernel 4: persistent single-workgroup LSTM (16 waves) + linear head.
//   Wave w owns hidden cols [16w,16w+16): gate N-tiles {w,16+w,32+w,48+w}.
//   A = [y_t | h] (K=320). h ping-pongs between two LDS fragment buffers so
//   only one barrier per step is needed. Bias enters as the C operand of the
//   ks=0 WMMA. Weights stream from L2-resident wpack as b128 loads.
// ---------------------------------------------------------------------------
__global__ void __launch_bounds__(512) k_lstm(
    const __bf16* __restrict__ yfrag, const __bf16* __restrict__ wpack,
    const float* __restrict__ hx0, const float* __restrict__ cx0,
    const float* __restrict__ bih, const float* __restrict__ bhh,
    const float* __restrict__ linw, const float* __restrict__ linb,
    float* __restrict__ out) {
    extern __shared__ char smem[];
    __bf16* hb0    = (__bf16*)smem;                  // 16384 bf16 = 32KB
    __bf16* hb1    = (__bf16*)(smem + 32768);        // 16384 bf16 = 32KB
    float*  hplain = (float*)(smem + 65536);         // 16384 f32  = 64KB

    int tid = threadIdx.x;
    int w = tid >> 5, L = tid & 31;
    int laneHi = (L >= 16), ln = L & 15;

    // hb0 <- hx0 (frag layout: local ks 0..7 == global K tile 2..9)
    for (int i = tid; i < 16384; i += 512) {
        int e = i & 15, lane = (i >> 4) & 31, mt = (i >> 9) & 3, ks = i >> 11;
        int K = (ks + 2) * 32 + kmap(e, lane >= 16);
        int hcol = K - 64;
        int b = mt * 16 + (lane & 15);
        hb0[i] = (__bf16)hx0[b * 256 + hcol];
    }

    // Bias tiles as WMMA C operands (loop-invariant, col = lane's gate column).
    v8f cbias[4];
    #pragma unroll
    for (int gt = 0; gt < 4; gt++) {
        int n = gt * 256 + w * 16 + ln;
        float bv = bih[n] + bhh[n];
        #pragma unroll
        for (int i = 0; i < 8; i++) cbias[gt][i] = bv;
    }
    float c[4][8];
    #pragma unroll
    for (int mt = 0; mt < 4; mt++)
        #pragma unroll
        for (int rg = 0; rg < 8; rg++) {
            int b = mt * 16 + rg + (laneHi ? 8 : 0);
            c[mt][rg] = cx0[b * 256 + w * 16 + ln];
        }

    // Precompute this lane's h-store position inside the h fragment buffer.
    int q    = ((w & 1) ? 16 : 0) + ln;       // (64+hcol) % 32
    int hiN  = (ln >= 8);
    int Kl   = q - (hiN ? 8 : 0);
    int eh   = (Kl >= 16) ? (8 + Kl - 16) : Kl;
    int ksh  = w >> 1;                         // local K tile of this hidden col
    int lane2b = hiN * 16 + (laneHi ? 8 : 0);  // + rg
    __syncthreads();

    const __bf16* wpw[4];
    #pragma unroll
    for (int gt = 0; gt < 4; gt++) wpw[gt] = wpack + (size_t)((gt * 16 + w) * 10) * 512;

    for (int t = 0; t < 1024; t++) {
        const __bf16* hcur = (t & 1) ? hb1 : hb0;
        __bf16*       hnxt = (t & 1) ? hb0 : hb1;
        v8f acc[4][4];

        #pragma unroll
        for (int ks = 0; ks < 10; ks++) {
            Frag a[4];
            if (ks < 2) {               // A cols 0..63: conv output y_t (global/L2)
                #pragma unroll
                for (int mt = 0; mt < 4; mt++) {
                    const uint4* src =
                        (const uint4*)(yfrag + ((((size_t)t * 2 + ks) * 4 + mt) * 32 + L) * 16);
                    a[mt].q[0] = src[0]; a[mt].q[1] = src[1];
                }
            } else {                    // A cols 64..319: recurrent h (LDS)
                #pragma unroll
                for (int mt = 0; mt < 4; mt++) {
                    const uint4* src = (const uint4*)(hcur + (((ks - 2) * 4 + mt) * 32 + L) * 16);
                    a[mt].q[0] = src[0]; a[mt].q[1] = src[1];
                }
            }
            #pragma unroll
            for (int gt = 0; gt < 4; gt++) {
                Frag wf;
                const uint4* src = (const uint4*)(wpw[gt] + (ks * 32 + L) * 16);
                wf.q[0] = src[0]; wf.q[1] = src[1];
                if (ks == 0) {
                    #pragma unroll
                    for (int mt = 0; mt < 4; mt++)
                        acc[gt][mt] = __builtin_amdgcn_wmma_f32_16x16x32_bf16(
                            false, a[mt].v, false, wf.v, (short)0, cbias[gt], false, false);
                } else {
                    #pragma unroll
                    for (int mt = 0; mt < 4; mt++)
                        acc[gt][mt] = __builtin_amdgcn_wmma_f32_16x16x32_bf16(
                            false, a[mt].v, false, wf.v, (short)0, acc[gt][mt], false, false);
                }
            }
        }

        // Warm L2->WGP path for next step's y_t fragments while WMMAs drain.
        if (t < 1023) {
            const __bf16* pf =
                yfrag + ((((size_t)(t + 1) * 2 + (w & 1)) * 4 + ((w >> 1) & 3)) * 32 + L) * 16;
            __builtin_prefetch(pf, 0, 1);
        }

        // Wave-local LSTM elementwise update (i,f,g,o all owned by this wave).
        float hv[4][8];
        #pragma unroll
        for (int mt = 0; mt < 4; mt++)
            #pragma unroll
            for (int rg = 0; rg < 8; rg++) {
                float ig = fast_sigmoid(acc[0][mt][rg]);
                float fg = fast_sigmoid(acc[1][mt][rg]);
                float gg = fast_tanh(acc[2][mt][rg]);
                float og = fast_sigmoid(acc[3][mt][rg]);
                float cn = fg * c[mt][rg] + ig * gg;
                c[mt][rg] = cn;
                float h = og * fast_tanh(cn);
                hv[mt][rg] = h;
                hnxt[((ksh * 4 + mt) * 32 + (lane2b + rg)) * 16 + eh] = (__bf16)h;
            }
        if (t == 1023) {
            #pragma unroll
            for (int mt = 0; mt < 4; mt++)
                #pragma unroll
                for (int rg = 0; rg < 8; rg++) {
                    int b = mt * 16 + rg + (laneHi ? 8 : 0);
                    hplain[b * 256 + w * 16 + ln] = hv[mt][rg];
                }
        }
        __syncthreads();                 // new h visible; old buffer now reusable
    }

    // Final linear head: out[b][j] = lin_b[j] + h[b] . lin_w[j]
    for (int idx = tid; idx < 640; idx += 512) {
        int b = idx / 10, j = idx % 10;
        float s = linb[j];
        const float* hr = hplain + b * 256;
        const float* wr = linw + j * 256;
        for (int hc = 0; hc < 256; hc++) s += hr[hc] * wr[hc];
        out[b * 10 + j] = s;
    }
}

// ---------------------------------------------------------------------------
extern "C" void kernel_launch(void* const* d_in, const int* in_sizes, int n_in,
                              void* d_out, int out_size, void* d_ws, size_t ws_size,
                              hipStream_t stream) {
    const float* input  = (const float*)d_in[0];
    // d_in[1] = "r" scratch input (unused)
    const float* hx0    = (const float*)d_in[2];
    const float* cx0    = (const float*)d_in[3];
    const float* conv_w = (const float*)d_in[4];
    const float* conv_b = (const float*)d_in[5];
    const float* w_ih   = (const float*)d_in[6];
    const float* b_ih   = (const float*)d_in[7];
    const float* w_hh   = (const float*)d_in[8];
    const float* b_hh   = (const float*)d_in[9];
    const float* lin_w  = (const float*)d_in[10];
    const float* lin_b  = (const float*)d_in[11];

    char* ws = (char*)d_ws;
    float*  invn     = (float*)ws;                  //  32 KB
    __bf16* convpack = (__bf16*)(ws + 0x8000);      //  48 KB
    __bf16* wpack    = (__bf16*)(ws + 0x14000);     // 640 KB
    __bf16* yfrag    = (__bf16*)(ws + 0xB4000);     //   8 MB

    k_norm<<<dim3(64), dim3(128, 8), 0, stream>>>(input, invn);
    k_pack_conv<<<(12 * 4 * 32 * 16 + 255) / 256, 256, 0, stream>>>(conv_w, convpack);
    k_pack_w<<<(64 * 10 * 32 * 16 + 255) / 256, 256, 0, stream>>>(w_ih, w_hh, wpack);
    k_conv<<<dim3(16, 64), 128, 0, stream>>>(input, invn, conv_b, convpack, yfrag);
    k_lstm<<<1, 512, 131072, stream>>>(yfrag, wpack, hx0, cx0, b_ih, b_hh,
                                       lin_w, lin_b, (float*)d_out);
}
